// VectorQuantizerEMA_19756849562003
// MI455X (gfx1250) — matrix operand order repaired and checked by reading, not compiled, not run on hardware
//
#include <hip/hip_runtime.h>

#define N_IN  131072
#define D_DIM 512
#define K_EMB 1024
#define BT_STRIDE 520   // padded ushort row stride -> conflict-free b128 LDS reads
#define TILE_USHORTS (16 * BT_STRIDE)

// Async global->LDS path (ASYNCcnt) if the toolchain exposes the gfx1250 builtins.
#define VQ_ASYNC 0
#if defined(__has_builtin)
#if __has_builtin(__builtin_amdgcn_global_load_async_to_lds_b128) && \
    __has_builtin(__builtin_amdgcn_s_wait_asynccnt)
#undef VQ_ASYNC
#define VQ_ASYNC 1
#endif
#endif

typedef __attribute__((ext_vector_type(16))) __bf16         v16bf;
typedef __attribute__((ext_vector_type(8)))  float          v8f;
typedef __attribute__((ext_vector_type(8)))  unsigned short u16x8;

// GCC-style int4 vector: matches the builtin's parameter type exactly
// ('__attribute__((__vector_size__(4 * sizeof(int)))) int').
typedef int v4i __attribute__((vector_size(16)));
typedef __attribute__((address_space(1))) v4i* gv4i_p;   // global (prints as __device__)
typedef __attribute__((address_space(3))) v4i* lv4i_p;   // LDS

union BFrag { v16bf v; u16x8 h[2]; };

// f32 -> bf16 bits, round-to-nearest-even
static __device__ __forceinline__ unsigned short f2bf(float f) {
  unsigned int u = __float_as_uint(f);
  u += 0x7fffu + ((u >> 16) & 1u);
  return (unsigned short)(u >> 16);
}

#if VQ_ASYNC
// Issue this thread's share (4 x 16B) of one 16-code bf16 tile as async
// global->LDS transfers (tracked by ASYNCcnt, overlaps the WMMA chain).
static __device__ __forceinline__ void stage_tile_async(
    const unsigned short* __restrict__ src, unsigned short* dst_lds, int tid) {
#pragma unroll
  for (int j = 0; j < 4; ++j) {
    const int u   = tid + j * 256;
    const int row = u >> 6;
    const int col = (u & 63) * 8;
    __builtin_amdgcn_global_load_async_to_lds_b128(
        (gv4i_p)(src + (size_t)row * D_DIM + col),
        (lv4i_p)(dst_lds + row * BT_STRIDE + col),
        0, 0);
  }
}
#endif

// ---------------------------------------------------------------------------
// 1) Codebook prep: bf16 copy + per-code squared norms
// ---------------------------------------------------------------------------
__global__ __launch_bounds__(256) void vq_prep_codebook(
    const float* __restrict__ emb, unsigned short* __restrict__ ebf,
    float* __restrict__ enorm2) {
  __shared__ float s[256];
  const int k = blockIdx.x, t = threadIdx.x;
  const float2 e = *(const float2*)(emb + (size_t)k * D_DIM + t * 2);
  ebf[(size_t)k * D_DIM + t * 2 + 0] = f2bf(e.x);
  ebf[(size_t)k * D_DIM + t * 2 + 1] = f2bf(e.y);
  s[t] = e.x * e.x + e.y * e.y;
  __syncthreads();
  for (int o = 128; o > 0; o >>= 1) { if (t < o) s[t] += s[t + o]; __syncthreads(); }
  if (t == 0) enorm2[k] = s[0];
}

// ---------------------------------------------------------------------------
// 2) Fused distance-GEMM + argmin.  256 threads = 8 waves, 128 rows / block.
//    A fragments live in VGPRs; 16-code bf16 tiles double-buffered in LDS
//    (async global->LDS when available); 16 chained v_wmma per tile.
// ---------------------------------------------------------------------------
#define COMPUTE_TILE(CUR, NT)                                                  \
  {                                                                            \
    v8f acc;                                                                   \
    _Pragma("unroll") for (int j = 0; j < 8; ++j) acc[j] = 0.0f;               \
    _Pragma("unroll") for (int kc = 0; kc < 16; ++kc) {                        \
      const int k0 = kc * 32 + ((lane & 16) ? 8 : 0);                          \
      BFrag b;                                                                 \
      b.h[0] = *(const u16x8*)((CUR) + m * BT_STRIDE + k0);                    \
      b.h[1] = *(const u16x8*)((CUR) + m * BT_STRIDE + k0 + 16);               \
      acc = __builtin_amdgcn_wmma_f32_16x16x32_bf16(                           \
          false, a[kc].v, false, b.v, (short)0, acc, false, false);            \
    }                                                                          \
    const int   code = (NT) * 16 + m;                                          \
    const float en   = enorm2[code];                                           \
    _Pragma("unroll") for (int r = 0; r < 8; ++r) {                            \
      const float dv = en - 2.0f * acc[r];                                     \
      if (dv < minv[r]) { minv[r] = dv; mini[r] = code; }                      \
    }                                                                          \
  }

__global__ __launch_bounds__(256) void vq_assign(
    const float* __restrict__ x, const unsigned short* __restrict__ ebf,
    const float* __restrict__ enorm2, int* __restrict__ idx_out,
    float* __restrict__ idxf_out) {
  __shared__ unsigned short bt[2 * TILE_USHORTS];
  struct Pair { float v; int i; };
  __shared__ Pair red[8][16][16];

  const int tid  = threadIdx.x;
  const int wave = tid >> 5;
  const int lane = tid & 31;
  const int m    = lane & 15;                          // A row / B column slot
  const int gr   = blockIdx.x * 128 + wave * 16 + m;   // this lane's input row

  // A fragments (16x32 bf16, ISA 7.12.2): lanes 0-15 hold K {0..7}U{16..23},
  // lanes 16-31 hold K {8..15}U{24..31}.
  BFrag a[16];
  #pragma unroll
  for (int kc = 0; kc < 16; ++kc) {
    const int k0 = kc * 32 + ((lane & 16) ? 8 : 0);
    const float* p = x + (size_t)gr * D_DIM + k0;
    const float4 f0 = *(const float4*)(p);
    const float4 f1 = *(const float4*)(p + 4);
    const float4 f2 = *(const float4*)(p + 16);
    const float4 f3 = *(const float4*)(p + 20);
    u16x8 h0, h1;
    h0[0]=f2bf(f0.x); h0[1]=f2bf(f0.y); h0[2]=f2bf(f0.z); h0[3]=f2bf(f0.w);
    h0[4]=f2bf(f1.x); h0[5]=f2bf(f1.y); h0[6]=f2bf(f1.z); h0[7]=f2bf(f1.w);
    h1[0]=f2bf(f2.x); h1[1]=f2bf(f2.y); h1[2]=f2bf(f2.z); h1[3]=f2bf(f2.w);
    h1[4]=f2bf(f3.x); h1[5]=f2bf(f3.y); h1[6]=f2bf(f3.z); h1[7]=f2bf(f3.w);
    a[kc].h[0] = h0; a[kc].h[1] = h1;
  }

  float minv[8];
  int   mini[8];
  #pragma unroll
  for (int r = 0; r < 8; ++r) { minv[r] = 3.4e38f; mini[r] = 0; }

#if VQ_ASYNC
  // Double-buffered: async-stage tile nt+1 while tile nt feeds the WMMA chain.
  stage_tile_async(ebf, bt, tid);
  for (int nt = 0; nt < 64; ++nt) {
    __builtin_amdgcn_s_wait_asynccnt(0);   // my share of tile nt landed
    __syncthreads();                       // everyone's share landed
    if (nt + 1 < 64)
      stage_tile_async(ebf + (size_t)(nt + 1) * 16 * D_DIM,
                       bt + ((nt + 1) & 1) * TILE_USHORTS, tid);
    const unsigned short* cur = bt + (nt & 1) * TILE_USHORTS;
    COMPUTE_TILE(cur, nt)
  }
#else
  // Fallback: synchronous staged copy, two barriers per tile.
  for (int nt = 0; nt < 64; ++nt) {
    __syncthreads();   // previous tile fully consumed
    #pragma unroll
    for (int j = 0; j < 4; ++j) {
      const int u   = tid + j * 256;
      const int row = u >> 6;
      const int col = (u & 63) * 8;
      *(u16x8*)(bt + row * BT_STRIDE + col) =
          *(const u16x8*)(ebf + (size_t)(nt * 16 + row) * D_DIM + col);
    }
    if (nt + 1 < 64)
      __builtin_prefetch(ebf + (size_t)(nt + 1) * 16 * D_DIM + (tid << 4), 0, 3);
    __syncthreads();
    const unsigned short* cur = bt;
    COMPUTE_TILE(cur, nt)
  }
#endif

  // Cross-lane argmin: D element (r,lane) is row r+(lane>=16?8:0), col m
  __syncthreads();
  const int mhi = (lane & 16) ? 8 : 0;
  #pragma unroll
  for (int r = 0; r < 8; ++r) {
    red[wave][r + mhi][m].v = minv[r];
    red[wave][r + mhi][m].i = mini[r];
  }
  __syncthreads();
  if (lane < 16) {
    float bv = red[wave][lane][0].v;
    int   bi = red[wave][lane][0].i;
    #pragma unroll
    for (int c = 1; c < 16; ++c) {
      const float v = red[wave][lane][c].v;
      const int   i = red[wave][lane][c].i;
      if (v < bv || (v == bv && i < bi)) { bv = v; bi = i; }
    }
    const int row = blockIdx.x * 128 + wave * 16 + lane;
    idx_out[row]  = bi;
    idxf_out[row] = (float)bi;
  }
}

// ---------------------------------------------------------------------------
// 3) Gather quantized, loss partials, counts, dw segment-sum.
//    One element per thread: fully coalesced b32 loads/stores/atomics.
//    Grid = N * 2 blocks; block b covers row b>>1, columns (b&1)*256 + tid.
// ---------------------------------------------------------------------------
__global__ __launch_bounds__(256) void vq_scatter(
    const float* __restrict__ x, const float* __restrict__ emb,
    const int* __restrict__ idx, float* __restrict__ q_out,
    float* __restrict__ dw, float* __restrict__ counts,
    float* __restrict__ loss_acc) {
  __shared__ float s[256];
  const int tid = threadIdx.x;
  const int r   = blockIdx.x >> 1;
  const int c   = ((blockIdx.x & 1) << 8) + tid;
  const int k   = idx[r];
  const float xv = x[(size_t)r * D_DIM + c];
  const float ev = emb[(size_t)k * D_DIM + c];
  q_out[(size_t)r * D_DIM + c] = ev;
  atomicAdd(dw + (size_t)k * D_DIM + c, xv);
  if (c == 0) atomicAdd(counts + k, 1.0f);
  const float d = ev - xv;
  s[tid] = d * d;
  __syncthreads();
  for (int o = 128; o > 0; o >>= 1) { if (tid < o) s[tid] += s[tid + o]; __syncthreads(); }
  if (tid == 0) atomicAdd(loss_acc, s[0]);
}

// ---------------------------------------------------------------------------
// 4) EMA cluster sizes + Laplace smoothing + loss scalar (single block)
// ---------------------------------------------------------------------------
__global__ __launch_bounds__(1024) void vq_finalize_cs(
    const float* __restrict__ ema_cs, const float* __restrict__ counts,
    const float* __restrict__ loss_acc, float* __restrict__ cs_out,
    float* __restrict__ inv, float* __restrict__ loss_out) {
  __shared__ float s[1024];
  __shared__ float ns;
  const int t = threadIdx.x;
  const float c = ema_cs[t] * 0.99f + 0.01f * counts[t];
  s[t] = c;
  __syncthreads();
  for (int o = 512; o > 0; o >>= 1) { if (t < o) s[t] += s[t + o]; __syncthreads(); }
  if (t == 0) ns = s[0];
  __syncthreads();
  const float n   = ns;
  const float ncs = (c + 1e-5f) / (n + (float)K_EMB * 1e-5f) * n;
  cs_out[t] = ncs;
  inv[t]    = 1.0f / ncs;
  if (t == 0)
    loss_out[0] = 0.25f * loss_acc[0] / ((float)N_IN * (float)D_DIM);
}

// ---------------------------------------------------------------------------
// 5) new_ema_w and new_embedding
// ---------------------------------------------------------------------------
__global__ __launch_bounds__(256) void vq_finalize_w(
    const float* __restrict__ ema_w, const float* __restrict__ dw,
    const float* __restrict__ inv, float* __restrict__ emaw_out,
    float* __restrict__ emb_out) {
  const int i = blockIdx.x * 256 + threadIdx.x;          // float4 index
  const float4 w = *(const float4*)(ema_w + (size_t)i * 4);
  const float4 d = *(const float4*)(dw + (size_t)i * 4);
  const int   k  = (i * 4) >> 9;                         // /512
  const float iv = inv[k];
  const float n0 = w.x * 0.99f + 0.01f * d.x;
  const float n1 = w.y * 0.99f + 0.01f * d.y;
  const float n2 = w.z * 0.99f + 0.01f * d.z;
  const float n3 = w.w * 0.99f + 0.01f * d.w;
  float* wp = emaw_out + (size_t)i * 4;   // odd global offset -> scalar stores
  float* ep = emb_out  + (size_t)i * 4;
  wp[0] = n0; wp[1] = n1; wp[2] = n2; wp[3] = n3;
  ep[0] = n0 * iv; ep[1] = n1 * iv; ep[2] = n2 * iv; ep[3] = n3 * iv;
}

// ---------------------------------------------------------------------------
extern "C" void kernel_launch(void* const* d_in, const int* in_sizes, int n_in,
                              void* d_out, int out_size, void* d_ws, size_t ws_size,
                              hipStream_t stream) {
  (void)in_sizes; (void)n_in; (void)out_size; (void)ws_size;
  const float* x      = (const float*)d_in[0];
  const float* emb    = (const float*)d_in[1];
  const float* ema_cs = (const float*)d_in[2];
  const float* ema_w  = (const float*)d_in[3];

  float* out      = (float*)d_out;
  float* loss_out = out;                                   // [1]
  float* q_out    = out + 1;                               // [N, D]
  float* idxf     = q_out + (size_t)N_IN * D_DIM;          // [N, 1]
  float* cs_out   = idxf + N_IN;                           // [K]
  float* emaw_out = cs_out + K_EMB;                        // [K, D]
  float* emb_out  = emaw_out + (size_t)K_EMB * D_DIM;      // [K, D]

  char* ws = (char*)d_ws;
  size_t cur = 0;
  auto take = [&](size_t bytes) { char* p = ws + cur; cur = (cur + bytes + 255) & ~(size_t)255; return p; };
  unsigned short* ebf      = (unsigned short*)take((size_t)K_EMB * D_DIM * 2); // 1 MB
  float*          enorm2   = (float*)take(K_EMB * 4);
  int*            idx      = (int*)take((size_t)N_IN * 4);                     // 512 KB
  float*          counts   = (float*)take(K_EMB * 4);
  float*          dw       = (float*)take((size_t)K_EMB * D_DIM * 4);          // 2 MB
  float*          loss_acc = (float*)take(4);
  float*          invcs    = (float*)take(K_EMB * 4);

  (void)hipMemsetAsync(counts, 0, K_EMB * 4, stream);
  (void)hipMemsetAsync(dw, 0, (size_t)K_EMB * D_DIM * 4, stream);
  (void)hipMemsetAsync(loss_acc, 0, 4, stream);

  vq_prep_codebook<<<K_EMB, 256, 0, stream>>>(emb, ebf, enorm2);
  vq_assign<<<N_IN / 128, 256, 0, stream>>>(x, ebf, enorm2, idx, idxf);
  vq_scatter<<<N_IN * 2, 256, 0, stream>>>(x, emb, idx, q_out, dw, counts, loss_acc);
  vq_finalize_cs<<<1, 1024, 0, stream>>>(ema_cs, counts, loss_acc, cs_out, invcs, loss_out);
  vq_finalize_w<<<(K_EMB * D_DIM / 4) / 256, 256, 0, stream>>>(ema_w, dw, invcs, emaw_out, emb_out);
}
